// SomaHeterogeneousSNN_2637109920457
// MI455X (gfx1250) — compile-verified, matrix operand order checked by reading, not run on hardware
//
#include <hip/hip_runtime.h>
#include <hip/hip_bf16.h>

typedef __attribute__((ext_vector_type(16))) _Float16 v16h;
typedef __attribute__((ext_vector_type(8)))  float    v8f;

#define B_   256
#define T_   1024
#define I_   64
#define H_   512
#define O_   10
#define MT   16

// ===========================================================================
// PASS 1: layer-1 LIF, barrier-free. One wave32 per (16-batch x 16-hidden)
// tile, serial over T with state in registers. grid = (B/16, H/16).
// ===========================================================================
__global__ __launch_bounds__(32)
void snn_layer1(const float* __restrict__ x,      // [B,T,I]
                const float* __restrict__ W1,     // [H,I]
                const float* __restrict__ b1,     // [H]
                const float* __restrict__ tau_m,  // [H]
                const float* __restrict__ v_th,   // [H]
                const float* __restrict__ v_rst,  // [H]
                const float* __restrict__ a_str,  // [H]
                float* __restrict__ hidden)       // [B,T,H]
{
    const int lane   = threadIdx.x & 31;
    const int laneLo = lane & 15;
    const int isHi   = lane >> 4;
    const int kb     = isHi * 8;                  // ISA 16-bit A/B lane K-base
    const int b0     = blockIdx.x * MT;
    const int hcol   = blockIdx.y * 16 + laneLo;  // this lane's hidden column

    // W1^T B-fragments (f16), loop-invariant in registers
    v16h W1f[2];
    #pragma unroll
    for (int c = 0; c < 2; ++c) {
        const float* p = W1 + hcol * I_ + c * 32 + kb;
        v16h f;
        #pragma unroll
        for (int e = 0; e < 8; ++e) { f[e] = (_Float16)p[e]; f[8+e] = (_Float16)p[16+e]; }
        W1f[c] = f;
    }
    const float dec = 1.f - 1.f / tau_m[hcol];
    const float vr  = v_rst[hcol];
    const float st  = a_str[hcol];     // >0 iff adapt_mask
    const float bb  = b1[hcol];
    const float vt0 = v_th[hcol];

    float v1s[8], vts[8];
    #pragma unroll
    for (int r = 0; r < 8; ++r) { v1s[r] = 0.f; vts[r] = vt0; }

    const size_t xrow = (size_t)(b0 + laneLo) * T_;

    for (int t = 0; t < T_; ++t) {
        const size_t xb = (xrow + t) * I_ + kb;
        const float* p0 = x + xb;
        const float* p1 = x + xb + 32;
        v16h A0, A1;
        #pragma unroll
        for (int e = 0; e < 8; ++e) {
            A0[e] = (_Float16)p0[e];  A0[8+e] = (_Float16)p0[16+e];
            A1[e] = (_Float16)p1[e];  A1[8+e] = (_Float16)p1[16+e];
        }
        if (t + 1 < T_) __builtin_prefetch(x + xb + I_, 0, 3);

        v8f acc = {};
        acc = __builtin_amdgcn_wmma_f32_16x16x32_f16(false, A0, false, W1f[0],
                                                     (short)0, acc, false, false);
        acc = __builtin_amdgcn_wmma_f32_16x16x32_f16(false, A1, false, W1f[1],
                                                     (short)0, acc, false, false);
        #pragma unroll
        for (int r = 0; r < 8; ++r) {
            float v   = v1s[r] * dec + acc[r] + bb;
            float vth = vts[r];
            const bool sp = (v - vth) >= 0.f;
            const float sf = sp ? 1.f : 0.f;
            if (sp) { v = vr; vth = (st > 0.f) ? vth + st : vth * 0.99f; }
            else    { vth = vth * 0.99f; }
            v1s[r] = v; vts[r] = vth;
            const int mr = r + isHi * 8;
            // streaming output: non-temporal store, keep L2 for x/W1
            __builtin_nontemporal_store(
                sf, &hidden[((size_t)(b0 + mr) * T_ + t) * H_ + hcol]);
        }
    }
}

// ===========================================================================
// PASS 2a: h2[b,t,o] = s1[b,t,:] @ W2^T + b2 — fully t-parallel GEMM.
// M = flattened (b,t) rows (16 per wave), K = 512 (16 chained WMMAs), N = 16.
// ===========================================================================
__global__ __launch_bounds__(256)
void snn_h2(const float* __restrict__ hidden,   // [B,T,H] (spikes, f32)
            const float* __restrict__ W2,       // [O,H]
            const float* __restrict__ b2,       // [O]
            float* __restrict__ h2all)          // [B,T,O] workspace
{
    __shared__ _Float16 wb[16][H_];             // W2 padded 10->16 rows, f16

    const int tid    = threadIdx.x;
    const int w      = tid >> 5;
    const int lane   = tid & 31;
    const int laneLo = lane & 15;
    const int isHi   = lane >> 4;
    const int kb     = isHi * 8;

    for (int i = tid; i < 16 * H_; i += 256) {
        const int n = i >> 9, k = i & (H_ - 1);
        wb[n][k] = (_Float16)((n < O_) ? W2[n * H_ + k] : 0.f);
    }
    __syncthreads();

    const int tile = blockIdx.x * 8 + w;        // 16384 M-tiles
    const int b  = tile >> 6;                   // T/16 = 64 tiles per batch row
    const int t0 = (tile & 63) << 4;

    const size_t rowbase = ((size_t)b * T_ + t0 + laneLo) * H_;
    v8f acc = {};
    #pragma unroll 4
    for (int kc = 0; kc < 16; ++kc) {
        const float* p = hidden + rowbase + kc * 32 + kb;
        v16h A, Bf;
        #pragma unroll
        for (int e = 0; e < 8; ++e) {           // spikes: read-once -> NT loads
            A[e]   = (_Float16)__builtin_nontemporal_load(p + e);
            A[8+e] = (_Float16)__builtin_nontemporal_load(p + 16 + e);
            Bf[e]   = wb[laneLo][kc * 32 + kb + e];
            Bf[8+e] = wb[laneLo][kc * 32 + kb + 16 + e];
        }
        acc = __builtin_amdgcn_wmma_f32_16x16x32_f16(false, A, false, Bf,
                                                     (short)0, acc, false, false);
    }
    if (laneLo < O_) {
        const float b2v = b2[laneLo];
        #pragma unroll
        for (int r = 0; r < 8; ++r) {
            const int m = r + isHi * 8;
            h2all[((size_t)b * T_ + t0 + m) * O_ + laneLo] = acc[r] + b2v;
        }
    }
}

// ===========================================================================
// PASS 2b: output-LIF scan, one thread per (b,o) chain.
// ===========================================================================
__global__ __launch_bounds__(256)
void snn_out_scan(const float* __restrict__ h2all, float* __restrict__ out)
{
    const int id = blockIdx.x * blockDim.x + threadIdx.x;
    if (id >= B_ * O_) return;
    const int b = id / O_, o = id - b * O_;
    const float* p = h2all + (size_t)b * T_ * O_ + o;
    float v2 = 0.f, acc = 0.f;
    for (int t = 0; t < T_; ++t) {
        const float h2 = p[(size_t)t * O_];
        v2 += (h2 - v2) * 0.1f;
        if ((v2 - 1.f) >= 0.f) { acc += 1.f; v2 = 0.f; }
    }
    out[id] = acc;
}

// ===========================================================================
// Fallback: fused persistent kernel (used only if workspace too small).
// ===========================================================================
__global__ __launch_bounds__(512)
void snn_fused(const float* __restrict__ x, const float* __restrict__ W1,
               const float* __restrict__ b1, const float* __restrict__ W2,
               const float* __restrict__ b2, const float* __restrict__ tau_m,
               const float* __restrict__ v_th, const float* __restrict__ v_rst,
               const float* __restrict__ a_str,
               float* __restrict__ out, float* __restrict__ hidden)
{
    __shared__ _Float16 s1buf[MT][H_];
    __shared__ float    h2red[MT * 16];
    __shared__ float    v2s  [MT * 16];
    __shared__ float    oacc [MT * 16];

    const int tid = threadIdx.x, w = tid >> 5, lane = tid & 31;
    const int laneLo = lane & 15, isHi = lane >> 4, kb = isHi * 8;
    const int b0 = blockIdx.x * MT;

    if (tid < 256) { h2red[tid] = 0.f; v2s[tid] = 0.f; oacc[tid] = 0.f; }

    v16h  W1f[2][2];
    float dec[2], vr[2], st[2], bb1[2], v1s[2][8], vts[2][8];
    #pragma unroll
    for (int j = 0; j < 2; ++j) {
        const int hrow = w * 32 + j * 16 + laneLo;
        #pragma unroll
        for (int c = 0; c < 2; ++c) {
            const float* p = W1 + hrow * I_ + c * 32 + kb;
            v16h f;
            #pragma unroll
            for (int e = 0; e < 8; ++e) { f[e] = (_Float16)p[e]; f[8+e] = (_Float16)p[16+e]; }
            W1f[j][c] = f;
        }
        dec[j] = 1.f - 1.f / tau_m[hrow];
        vr[j] = v_rst[hrow]; st[j] = a_str[hrow]; bb1[j] = b1[hrow];
        const float vt0 = v_th[hrow];
        #pragma unroll
        for (int r = 0; r < 8; ++r) { v1s[j][r] = 0.f; vts[j][r] = vt0; }
    }
    v16h W2f;
    if (laneLo < O_) {
        const float* p = W2 + laneLo * H_ + w * 32 + kb;
        #pragma unroll
        for (int e = 0; e < 8; ++e) { W2f[e] = (_Float16)p[e]; W2f[8+e] = (_Float16)p[16+e]; }
    } else {
        #pragma unroll
        for (int e = 0; e < 16; ++e) W2f[e] = (_Float16)0.f;
    }
    float b2v = 0.f;
    if (tid < 256 && (tid & 15) < O_) b2v = b2[tid & 15];
    __syncthreads();

    const size_t xrow = (size_t)(b0 + laneLo) * T_;
    for (int t = 0; t < T_; ++t) {
        const size_t xb = (xrow + t) * I_ + kb;
        const float* p0 = x + xb; const float* p1 = x + xb + 32;
        v16h A0, A1;
        #pragma unroll
        for (int e = 0; e < 8; ++e) {
            A0[e] = (_Float16)p0[e]; A0[8+e] = (_Float16)p0[16+e];
            A1[e] = (_Float16)p1[e]; A1[8+e] = (_Float16)p1[16+e];
        }
        #pragma unroll
        for (int j = 0; j < 2; ++j) {
            v8f acc = {};
            acc = __builtin_amdgcn_wmma_f32_16x16x32_f16(false, A0, false, W1f[j][0],
                                                         (short)0, acc, false, false);
            acc = __builtin_amdgcn_wmma_f32_16x16x32_f16(false, A1, false, W1f[j][1],
                                                         (short)0, acc, false, false);
            const int hcol = w * 32 + j * 16 + laneLo;
            #pragma unroll
            for (int r = 0; r < 8; ++r) {
                float v = v1s[j][r] * dec[j] + acc[r] + bb1[j];
                float vth = vts[j][r];
                const bool sp = (v - vth) >= 0.f;
                const float sf = sp ? 1.f : 0.f;
                if (sp) { v = vr[j]; vth = (st[j] > 0.f) ? vth + st[j] : vth * 0.99f; }
                else    { vth = vth * 0.99f; }
                v1s[j][r] = v; vts[j][r] = vth;
                const int mr = r + isHi * 8;
                s1buf[mr][hcol] = (_Float16)sf;
                __builtin_nontemporal_store(
                    sf, &hidden[((size_t)(b0 + mr) * T_ + t) * H_ + hcol]);
            }
        }
        __syncthreads();
        v16h As;
        {
            const int ks = w * 32 + kb;
            #pragma unroll
            for (int e = 0; e < 8; ++e) {
                As[e] = s1buf[laneLo][ks + e]; As[8+e] = s1buf[laneLo][ks + 16 + e];
            }
        }
        v8f zc = {};
        v8f part = __builtin_amdgcn_wmma_f32_16x16x32_f16(false, As, false, W2f,
                                                          (short)0, zc, false, false);
        #pragma unroll
        for (int r = 0; r < 8; ++r)
            atomicAdd(&h2red[(r + isHi * 8) * 16 + laneLo], part[r]);
        __syncthreads();
        if (tid < 256) {
            const int n = tid & 15;
            if (n < O_) {
                const float h2 = h2red[tid] + b2v;
                float v2 = v2s[tid];
                v2 += (h2 - v2) * 0.1f;
                if ((v2 - 1.f) >= 0.f) { oacc[tid] += 1.f; v2 = 0.f; }
                v2s[tid] = v2;
            }
            h2red[tid] = 0.f;
        }
        __syncthreads();
    }
    if (tid < 256) {
        const int m = tid >> 4, n = tid & 15;
        if (n < O_) out[(size_t)(b0 + m) * O_ + n] = oacc[tid];
    }
}

extern "C" void kernel_launch(void* const* d_in, const int* in_sizes, int n_in,
                              void* d_out, int out_size, void* d_ws, size_t ws_size,
                              hipStream_t stream) {
    const float* x     = (const float*)d_in[0];
    const float* W1    = (const float*)d_in[1];
    const float* b1    = (const float*)d_in[2];
    const float* W2    = (const float*)d_in[3];
    const float* b2    = (const float*)d_in[4];
    const float* tau_m = (const float*)d_in[5];
    const float* v_th  = (const float*)d_in[6];
    const float* v_rst = (const float*)d_in[7];
    const float* a_str = (const float*)d_in[9];   // d_in[8]=adapt_mask, derivable

    float* out    = (float*)d_out;                          // [B,O]
    float* hidden = (float*)d_out + (size_t)B_ * O_;        // [B,T,H]

    const size_t need = (size_t)B_ * T_ * O_ * sizeof(float);   // 10 MB
    if (ws_size >= need) {
        float* h2all = (float*)d_ws;
        snn_layer1<<<dim3(B_ / MT, H_ / 16), 32, 0, stream>>>(
            x, W1, b1, tau_m, v_th, v_rst, a_str, hidden);
        snn_h2<<<(B_ * (T_ / 16)) / 8, 256, 0, stream>>>(hidden, W2, b2, h2all);
        snn_out_scan<<<(B_ * O_ + 255) / 256, 256, 0, stream>>>(h2all, out);
    } else {
        snn_fused<<<B_ / MT, 512, 0, stream>>>(x, W1, b1, W2, b2, tau_m, v_th,
                                               v_rst, a_str, out, hidden);
    }
}
